// SwitchHeadSelfAttention_23974507446595
// MI455X (gfx1250) — matrix operand order repaired and checked by reading, not compiled, not run on hardware
//
#include <hip/hip_runtime.h>
#include <hip/hip_bf16.h>

#define D_MODEL 1024
#define N_HEADS 8
#define D_HEAD 128
#define N_EXPERTS 4
#define BATCH 8
#define SEQ 1024
#define NTOK (BATCH*SEQ)
#define LN_EPS 1e-5f

typedef __attribute__((ext_vector_type(16))) __bf16 v16bf;
typedef __attribute__((ext_vector_type(8)))  float v8f;
typedef __attribute__((ext_vector_type(4)))  unsigned int u32x4;
typedef __attribute__((ext_vector_type(8)))  int i32x8;
typedef __attribute__((ext_vector_type(4)))  int i32x4;

#if defined(__gfx1250__) && __has_builtin(__builtin_amdgcn_tensor_load_to_lds) && __has_builtin(__builtin_amdgcn_s_wait_tensorcnt)
#define USE_TDM 1
#else
#define USE_TDM 0
#endif

union BF16x16 {
  v16bf v;
  uint4 q[2];
  unsigned short s[16];
};

__device__ __forceinline__ unsigned short f2bf(float f) {
  union { float f; unsigned u; } v; v.f = f;
  unsigned r = (v.u + 0x7FFFu + ((v.u >> 16) & 1u)) >> 16;   // round-to-nearest-even
  return (unsigned short)r;
}
__device__ __forceinline__ float bf2f(unsigned short s) {
  union { unsigned u; float f; } v; v.u = ((unsigned)s) << 16;
  return v.f;
}

// Load a 16-element bf16 fragment from two 16B-aligned addresses.
// A-matrix 16x32: lane holds row=lane%16, e[0..7]=K k0..k0+7, e[8..15]=K k0+16..k0+23 (k0=(lane/16)*8)
// B-matrix 32x16 (stored as B^T[N][K]): lane holds col N=lane%16, same K order as A
__device__ __forceinline__ v16bf load_frag(const unsigned short* p0, const unsigned short* p1) {
  BF16x16 u;
  u.q[0] = *(const uint4*)p0;
  u.q[1] = *(const uint4*)p1;
  return u.v;
}

__device__ __forceinline__ v8f wmma_bf16(v16bf a, v16bf b, v8f c) {
  return __builtin_amdgcn_wmma_f32_16x16x32_bf16(false, a, false, b, (short)0, c, false, false);
}

__device__ __forceinline__ v8f zero_v8f() {
  v8f z;
  #pragma unroll
  for (int i = 0; i < 8; ++i) z[i] = 0.f;
  return z;
}

// ---------------------------------------------------------------------------
// TDM: DMA a 128-row x 32-bf16 (64B/row) weight chunk into LDS.
// D# per CDNA5 ISA: group0 = {count/flags, lds_addr,
// global_addr, type=2}; group1 = {data_size=8B, tensor_dim0=8u, tensor_dim1=128,
// tile_dim0=8u, tile_dim1=128, dim0_stride=256u (2048B row pitch)}.
// ---------------------------------------------------------------------------
__device__ __forceinline__ void tdm_load_tile_128x32(unsigned lds_byte,
                                                     unsigned long long gaddr) {
#if USE_TDM
  u32x4 g0;
  g0[0] = 1u;                                            // count=1, user mode
  g0[1] = lds_byte;                                      // LDS byte address
  g0[2] = (unsigned)(gaddr & 0xFFFFFFFFull);             // global_addr[31:0]
  g0[3] = (unsigned)((gaddr >> 32) & 0x01FFFFFFull) | (2u << 30);  // addr msbs | type=2
  i32x8 g1;
  g1[0] = (3 << 16);        // workgroup_mask=0, data_size=3 (8B)
  g1[1] = (8 << 16);        // tensor_dim0 = 8 units (64B)
  g1[2] = (128 << 16);      // tensor_dim1 = 128 rows
  g1[3] = (8 << 16);        // tile_dim0 = 8 units
  g1[4] = 128;              // tile_dim1 = 128, tile_dim2 = 0
  g1[5] = 256;              // tensor_dim0_stride = 256 units (2048B)
  g1[6] = 0;
  g1[7] = 0;                // tensor_dim1_stride = 0 (2D)
  i32x4 gz = {0, 0, 0, 0};
#if __clang_major__ >= 23
  i32x8 gz8 = {0, 0, 0, 0, 0, 0, 0, 0};
  __builtin_amdgcn_tensor_load_to_lds(g0, g1, gz, gz, gz8, 0);
#else
  __builtin_amdgcn_tensor_load_to_lds(g0, g1, gz, gz, 0);
#endif
#endif
}

// ---------------------------------------------------------------------------
// 1) Weight convert: W[g][K][N] (f32) -> WT[g][N][K] (bf16), output-coalesced
// ---------------------------------------------------------------------------
__global__ void cvt_kernel(const float* __restrict__ W, unsigned short* __restrict__ WT,
                           int G, int K, int N) {
  size_t idx = (size_t)blockIdx.x * blockDim.x + threadIdx.x;
  size_t total = (size_t)G * K * N;
  if (idx >= total) return;
  int k = (int)(idx % K);
  size_t t = idx / K;
  int n = (int)(t % N);
  int g = (int)(t / N);
  WT[idx] = f2bf(W[((size_t)g * K + k) * N + n]);
}

// ---------------------------------------------------------------------------
// 2) LayerNorm -> bf16 xn. One wave per token, 8 tokens per 256-thread block.
// ---------------------------------------------------------------------------
__global__ void ln_kernel(const float* __restrict__ x, const float* __restrict__ g,
                          const float* __restrict__ b, unsigned short* __restrict__ xn) {
  int w = threadIdx.x >> 5, lane = threadIdx.x & 31;
  int t = blockIdx.x * 8 + w;
  const float* row = x + (size_t)t * D_MODEL;
  float s = 0.f, ss = 0.f;
  #pragma unroll 8
  for (int i = 0; i < 32; ++i) { float v = row[lane + i * 32]; s += v; ss += v * v; }
  #pragma unroll
  for (int m = 1; m < 32; m <<= 1) { s += __shfl_xor(s, m, 32); ss += __shfl_xor(ss, m, 32); }
  float mu = s * (1.f / D_MODEL);
  float var = ss * (1.f / D_MODEL) - mu * mu;
  float rs = rsqrtf(var + LN_EPS);
  unsigned short* orow = xn + (size_t)t * D_MODEL;
  #pragma unroll 8
  for (int i = 0; i < 32; ++i) {
    int d = lane + i * 32;
    orow[d] = f2bf((row[d] - mu) * rs * g[d] + b[d]);
  }
}

// ---------------------------------------------------------------------------
// 3) Gates: sigmoid(xn @ Wsel) then top-2 over E=4 per head, dense scatter.
// ---------------------------------------------------------------------------
__device__ __forceinline__ void top2_write(const float* a, float* out) {
  int i1 = 0; float v1 = a[0];
  #pragma unroll
  for (int e = 1; e < 4; ++e) if (a[e] > v1) { v1 = a[e]; i1 = e; }
  int i2 = -1; float v2 = -1e30f;
  #pragma unroll
  for (int e = 0; e < 4; ++e) if (e != i1 && a[e] > v2) { v2 = a[e]; i2 = e; }
  #pragma unroll
  for (int e = 0; e < 4; ++e) out[e] = (e == i1) ? v1 : ((e == i2) ? v2 : 0.f);
}

__global__ void gate_kernel(const unsigned short* __restrict__ xn,
                            const float* __restrict__ Wsv, const float* __restrict__ Wso,
                            float* __restrict__ gv, float* __restrict__ go) {
  int w = threadIdx.x >> 5, lane = threadIdx.x & 31;
  int t = blockIdx.x * 8 + w;
  const unsigned short* row = xn + (size_t)t * D_MODEL;
  float av = 0.f, ao = 0.f;
  #pragma unroll 4
  for (int d = 0; d < D_MODEL; ++d) {
    float xv = bf2f(row[d]);
    av += xv * Wsv[(size_t)d * 32 + lane];
    ao += xv * Wso[(size_t)d * 32 + lane];
  }
  float sv = 1.f / (1.f + __expf(-av));
  float so = 1.f / (1.f + __expf(-ao));
  __shared__ float shv[8][32], sho[8][32];
  shv[w][lane] = sv; sho[w][lane] = so;
  __syncthreads();
  if (lane < 8) {
    int h = lane;
    float a[4], o4[4];
    #pragma unroll
    for (int e = 0; e < 4; ++e) a[e] = shv[w][h * 4 + e];
    top2_write(a, o4);
    #pragma unroll
    for (int e = 0; e < 4; ++e) gv[(size_t)t * 32 + h * 4 + e] = o4[e];
    #pragma unroll
    for (int e = 0; e < 4; ++e) a[e] = sho[w][h * 4 + e];
    top2_write(a, o4);
    #pragma unroll
    for (int e = 0; e < 4; ++e) go[(size_t)t * 32 + h * 4 + e] = o4[e];
  }
}

// ---------------------------------------------------------------------------
// 4) Q/K projection + scale + RoPE. 4 waves/block, each wave 16 tokens x 128
//    cols. Weight chunk (128x32 bf16) staged in LDS by TDM, double-buffered:
//    DMA of chunk k+1 overlaps WMMA of chunk k; synced via TENSORcnt+barrier.
//    blockIdx = (token group of 64, head, 0:q / 1:k)
// ---------------------------------------------------------------------------
__global__ void __launch_bounds__(128)
qk_proj_kernel(const unsigned short* __restrict__ xn,
               const unsigned short* __restrict__ WqT,
               const unsigned short* __restrict__ WkT,
               unsigned short* __restrict__ qout,
               unsigned short* __restrict__ kout) {
  __shared__ alignas(16) unsigned short wtile[2][128 * 32];  // 2 x 8KB
  const int tid = threadIdx.x;
  const int wave = tid >> 5;
  const int lane = tid & 31;
  const int lr = lane & 15, hi = lane >> 4;
  const int t0 = blockIdx.x * 64 + wave * 16;
  const int h = blockIdx.y;
  const bool is_q = (blockIdx.z == 0);
  const unsigned short* WT = is_q ? WqT : WkT;
  unsigned short* out = is_q ? qout : kout;
  const unsigned short* wbase = WT + (size_t)h * D_HEAD * D_MODEL;  // [128][1024]

  v8f acc[8];
  #pragma unroll
  for (int n = 0; n < 8; ++n) acc[n] = zero_v8f();

  const unsigned short* arow = xn + (size_t)(t0 + lr) * D_MODEL;

#if USE_TDM
  if (wave == 0)
    tdm_load_tile_128x32((unsigned)(size_t)&wtile[0][0],
                         (unsigned long long)(size_t)wbase);
#endif

  for (int kc = 0; kc < D_MODEL; kc += 32) {
    const int cur = (kc >> 5) & 1;
#if USE_TDM
    if (wave == 0) __builtin_amdgcn_s_wait_tensorcnt(0);   // chunk 'cur' landed
    __syncthreads();                                       // publish to all waves
    if (wave == 0 && kc + 32 < D_MODEL)                    // overlap next DMA
      tdm_load_tile_128x32((unsigned)(size_t)&wtile[cur ^ 1][0],
                           (unsigned long long)(size_t)(wbase + kc + 32));
#else
    __syncthreads();
    #pragma unroll
    for (int i = 0; i < 4; ++i) {                          // cooperative stage
      int idx = tid + i * 128;                             // 0..511 x 16B
      int row = idx >> 2, seg = idx & 3;
      *(uint4*)&wtile[cur][row * 32 + seg * 8] =
          *(const uint4*)(wbase + (size_t)row * D_MODEL + kc + seg * 8);
    }
    __syncthreads();
#endif
    __builtin_prefetch((const void*)(arow + kc + 64), 0, 1);
    v16bf a = load_frag(arow + kc + hi * 8, arow + kc + hi * 8 + 16);
    #pragma unroll
    for (int n = 0; n < 8; ++n) {
      const unsigned short* bp = &wtile[cur][(n * 16 + lr) * 32 + hi * 8];
      acc[n] = wmma_bf16(a, load_frag(bp, bp + 16), acc[n]);
    }
#if !USE_TDM
#else
#endif
  }

  if (is_q) {
    const float sc = 0.08838834764831845f;  // 128^-0.5
    #pragma unroll
    for (int n = 0; n < 8; ++n)
      #pragma unroll
      for (int r = 0; r < 8; ++r) acc[n][r] *= sc;
  }

  // RoPE: pair (col j, col j+64), j = n*16 + lr for n<4; pos = seq index of row.
  const int srow0 = (t0 % SEQ) + hi * 8;
  #pragma unroll
  for (int n = 0; n < 4; ++n) {
    int j = n * 16 + lr;
    float invf = __expf(-(float)j * (9.210340371976184f / 64.f));  // 10000^(-j/64)
    #pragma unroll
    for (int r = 0; r < 8; ++r) {
      float ang = (float)(srow0 + r) * invf;
      float sn, cs;
      __sincosf(ang, &sn, &cs);
      float lo = acc[n][r], hv = acc[n + 4][r];
      acc[n][r]     = lo * cs - hv * sn;
      acc[n + 4][r] = lo * sn + hv * cs;
    }
  }

  // store [b][h][s][dh]
  int bidx = t0 / SEQ, s0 = t0 % SEQ;
  unsigned short* obase = out + ((size_t)bidx * N_HEADS + h) * SEQ * D_HEAD;
  #pragma unroll
  for (int n = 0; n < 8; ++n)
    #pragma unroll
    for (int r = 0; r < 8; ++r)
      obase[(size_t)(s0 + r + hi * 8) * D_HEAD + n * 16 + lr] = f2bf(acc[n][r]);
}

// ---------------------------------------------------------------------------
// 5) V projection with gate folded into A rows; store v transposed [b][h][dh][s]
// ---------------------------------------------------------------------------
__global__ void __launch_bounds__(32)
v_proj_kernel(const unsigned short* __restrict__ xn,
              const unsigned short* __restrict__ WvT,
              const float* __restrict__ gate_v,
              unsigned short* __restrict__ vt) {
  const int lane = threadIdx.x & 31;
  const int lr = lane & 15, hi = lane >> 4;
  const int t0 = blockIdx.x * 16;
  const int h = blockIdx.y;

  v8f acc[8];
  #pragma unroll
  for (int n = 0; n < 8; ++n) acc[n] = zero_v8f();

  const unsigned short* arow = xn + (size_t)(t0 + lr) * D_MODEL;

  for (int e = 0; e < N_EXPERTS; ++e) {
    float g = gate_v[(size_t)(t0 + lr) * 32 + h * 4 + e];
    const unsigned short* wbase = WvT + ((size_t)(h * 4 + e) * D_HEAD + lr) * D_MODEL;
    for (int kc = 0; kc < D_MODEL; kc += 32) {
      __builtin_prefetch((const void*)(arow + kc + 64), 0, 1);
      BF16x16 ua;
      const unsigned short* ap = arow + kc + hi * 8;
      ua.q[0] = *(const uint4*)ap;
      ua.q[1] = *(const uint4*)(ap + 16);
      #pragma unroll
      for (int i = 0; i < 16; ++i) ua.s[i] = f2bf(bf2f(ua.s[i]) * g);  // g*(xn@W) == (g*xn)@W
      #pragma unroll
      for (int n = 0; n < 8; ++n) {
        const unsigned short* bp = wbase + (size_t)(n * 16) * D_MODEL + kc + hi * 8;
        acc[n] = wmma_bf16(ua.v, load_frag(bp, bp + 16), acc[n]);
      }
    }
  }

  int bidx = t0 / SEQ, s0 = t0 % SEQ;
  unsigned short* vb = vt + ((size_t)bidx * N_HEADS + h) * D_HEAD * SEQ;
  #pragma unroll
  for (int n = 0; n < 8; ++n)
    #pragma unroll
    for (int r = 0; r < 8; ++r)
      vb[(size_t)(n * 16 + lr) * SEQ + s0 + r + hi * 8] = f2bf(acc[n][r]);
}

// ---------------------------------------------------------------------------
// 6) Flash attention (causal). Wave = (b, h, 16 query rows). kv chunks of 32.
// ---------------------------------------------------------------------------
__global__ void __launch_bounds__(32)
attn_kernel(const unsigned short* __restrict__ q,
            const unsigned short* __restrict__ k,
            const unsigned short* __restrict__ vt,
            unsigned short* __restrict__ o) {
  __shared__ alignas(16) unsigned short P[16 * 32];  // probs transpose staging
  const int lane = threadIdx.x & 31;
  const int lr = lane & 15, hi = lane >> 4;
  const int q0 = blockIdx.x * 16;
  const int h = blockIdx.y;
  const int b = blockIdx.z;

  const unsigned short* qb = q + ((size_t)b * N_HEADS + h) * SEQ * D_HEAD;
  const unsigned short* kb = k + ((size_t)b * N_HEADS + h) * SEQ * D_HEAD;
  const unsigned short* vb = vt + ((size_t)b * N_HEADS + h) * D_HEAD * SEQ;

  v16bf aq[4];
  const unsigned short* qrow = qb + (size_t)(q0 + lr) * D_HEAD;
  #pragma unroll
  for (int c = 0; c < 4; ++c)
    aq[c] = load_frag(qrow + c * 32 + hi * 8, qrow + c * 32 + hi * 8 + 16);

  v8f acc[8];
  #pragma unroll
  for (int n = 0; n < 8; ++n) acc[n] = zero_v8f();
  float mrow[8], lrow[8];
  #pragma unroll
  for (int r = 0; r < 8; ++r) { mrow[r] = -1e30f; lrow[r] = 0.f; }

  for (int tb = 0; tb < q0 + 16; tb += 32) {
    // scores: two 16-col tiles for kv [tb, tb+32)
    v8f sc0 = zero_v8f(), sc1 = zero_v8f();
    #pragma unroll
    for (int c = 0; c < 4; ++c) {
      const unsigned short* k0p = kb + (size_t)(tb + lr) * D_HEAD + c * 32 + hi * 8;
      const unsigned short* k1p = kb + (size_t)(tb + 16 + lr) * D_HEAD + c * 32 + hi * 8;
      sc0 = wmma_bf16(aq[c], load_frag(k0p, k0p + 16), sc0);
      sc1 = wmma_bf16(aq[c], load_frag(k1p, k1p + 16), sc1);
    }
    // causal mask
    const int c0 = tb + lr, c1 = tb + 16 + lr;
    #pragma unroll
    for (int r = 0; r < 8; ++r) {
      int srow = q0 + r + hi * 8;
      if (c0 > srow) sc0[r] = -1e30f;
      if (c1 > srow) sc1[r] = -1e30f;
    }
    // online softmax per row
    #pragma unroll
    for (int r = 0; r < 8; ++r) {
      float rm = fmaxf(sc0[r], sc1[r]);
      #pragma unroll
      for (int msk = 1; msk < 16; msk <<= 1) rm = fmaxf(rm, __shfl_xor(rm, msk, 32));
      float mnew = fmaxf(mrow[r], rm);
      float alpha = __expf(mrow[r] - mnew);
      float p0 = __expf(sc0[r] - mnew);
      float p1 = __expf(sc1[r] - mnew);
      float ps = p0 + p1;
      #pragma unroll
      for (int msk = 1; msk < 16; msk <<= 1) ps += __shfl_xor(ps, msk, 32);
      lrow[r] = lrow[r] * alpha + ps;
      mrow[r] = mnew;
      #pragma unroll
      for (int n = 0; n < 8; ++n) acc[n][r] *= alpha;
      P[(r + hi * 8) * 32 + lr]      = f2bf(p0);
      P[(r + hi * 8) * 32 + 16 + lr] = f2bf(p1);
    }
    __syncthreads();
    // P (C layout) -> A layout via LDS, then acc += P @ V (v transposed = B^T)
    v16bf ap = load_frag(&P[lr * 32 + hi * 8], &P[lr * 32 + hi * 8 + 16]);
    #pragma unroll
    for (int n = 0; n < 8; ++n) {
      const unsigned short* vp = vb + (size_t)(n * 16 + lr) * SEQ + tb + hi * 8;
      acc[n] = wmma_bf16(ap, load_frag(vp, vp + 16), acc[n]);
    }
    __syncthreads();
  }

  // normalize + store o[b][s][h][dh]
  #pragma unroll
  for (int r = 0; r < 8; ++r) {
    float inv = 1.f / lrow[r];
    size_t rowoff = ((size_t)(b * SEQ + q0 + r + hi * 8) * N_HEADS + h) * (size_t)D_HEAD;
    #pragma unroll
    for (int n = 0; n < 8; ++n)
      o[rowoff + n * 16 + lr] = f2bf(acc[n][r] * inv);
  }
}

// ---------------------------------------------------------------------------
// 7) Output projection (gate_o folded into A rows) + residual. K = H*E*dh = 4096.
//    Wave = 16 tokens x 128 out-cols; blockIdx.y selects 128-col group.
// ---------------------------------------------------------------------------
__global__ void __launch_bounds__(32)
out_proj_kernel(const unsigned short* __restrict__ o,
                const unsigned short* __restrict__ WoT,
                const float* __restrict__ gate_o,
                const float* __restrict__ x,
                float* __restrict__ out) {
  const int lane = threadIdx.x & 31;
  const int lr = lane & 15, hi = lane >> 4;
  const int t0 = blockIdx.x * 16;
  const int d0 = blockIdx.y * 128;

  v8f acc[8];
  #pragma unroll
  for (int n = 0; n < 8; ++n) acc[n] = zero_v8f();

  const unsigned short* arow = o + (size_t)(t0 + lr) * (N_HEADS * D_HEAD);

  for (int h = 0; h < N_HEADS; ++h) {
    for (int e = 0; e < N_EXPERTS; ++e) {
      float g = gate_o[(size_t)(t0 + lr) * 32 + h * 4 + e];
      const unsigned short* wbase = WoT + ((size_t)(h * 4 + e) * D_MODEL + d0 + lr) * D_HEAD;
      #pragma unroll
      for (int kc = 0; kc < D_HEAD; kc += 32) {
        BF16x16 ua;
        const unsigned short* ap = arow + h * D_HEAD + kc + hi * 8;
        ua.q[0] = *(const uint4*)ap;
        ua.q[1] = *(const uint4*)(ap + 16);
        #pragma unroll
        for (int i = 0; i < 16; ++i) ua.s[i] = f2bf(bf2f(ua.s[i]) * g);
        #pragma unroll
        for (int n = 0; n < 8; ++n) {
          const unsigned short* bp = wbase + (size_t)(n * 16) * D_HEAD + kc + hi * 8;
          acc[n] = wmma_bf16(ua.v, load_frag(bp, bp + 16), acc[n]);
        }
      }
    }
  }

  #pragma unroll
  for (int r = 0; r < 8; ++r) {
    size_t tok = (size_t)(t0 + r + hi * 8);
    #pragma unroll
    for (int n = 0; n < 8; ++n) {
      size_t idx = tok * D_MODEL + d0 + n * 16 + lr;
      out[idx] = x[idx] + acc[n][r];
    }
  }
}

// ---------------------------------------------------------------------------
extern "C" void kernel_launch(void* const* d_in, const int* in_sizes, int n_in,
                              void* d_out, int out_size, void* d_ws, size_t ws_size,
                              hipStream_t stream) {
  (void)in_sizes; (void)n_in; (void)out_size; (void)ws_size;
  const float* x   = (const float*)d_in[0];
  const float* lng = (const float*)d_in[1];
  const float* lnb = (const float*)d_in[2];
  const float* Wq  = (const float*)d_in[3];
  const float* Wk  = (const float*)d_in[4];
  const float* Wv  = (const float*)d_in[5];
  const float* Wo  = (const float*)d_in[6];
  const float* Wsv = (const float*)d_in[7];
  const float* Wso = (const float*)d_in[8];
  float* out = (float*)d_out;

  char* base = (char*)d_ws;
  size_t off = 0;
  auto alloc = [&](size_t bytes) -> void* {
    void* r = base + off;
    off += (bytes + 255) & ~(size_t)255;
    return r;
  };

  const size_t qkv_e = (size_t)NTOK * N_HEADS * D_HEAD;  // 8.39M elems
  unsigned short* xn  = (unsigned short*)alloc((size_t)NTOK * D_MODEL * 2);
  unsigned short* qbf = (unsigned short*)alloc(qkv_e * 2);
  unsigned short* kbf = (unsigned short*)alloc(qkv_e * 2);
  unsigned short* vtb = (unsigned short*)alloc(qkv_e * 2);
  unsigned short* obf = (unsigned short*)alloc(qkv_e * 2);
  unsigned short* WqT = (unsigned short*)alloc((size_t)N_HEADS * D_HEAD * D_MODEL * 2);
  unsigned short* WkT = (unsigned short*)alloc((size_t)N_HEADS * D_HEAD * D_MODEL * 2);
  unsigned short* WvT = (unsigned short*)alloc((size_t)N_HEADS * N_EXPERTS * D_HEAD * D_MODEL * 2);
  unsigned short* WoT = (unsigned short*)alloc((size_t)N_HEADS * N_EXPERTS * D_MODEL * D_HEAD * 2);
  float* gv = (float*)alloc((size_t)NTOK * 32 * 4);
  float* go = (float*)alloc((size_t)NTOK * 32 * 4);

  // weight conversion (transposed bf16)
  {
    size_t n1 = (size_t)N_HEADS * D_MODEL * D_HEAD;
    size_t n2 = (size_t)N_HEADS * N_EXPERTS * D_MODEL * D_HEAD;
    cvt_kernel<<<(unsigned)((n1 + 255) / 256), 256, 0, stream>>>(Wq, WqT, N_HEADS, D_MODEL, D_HEAD);
    cvt_kernel<<<(unsigned)((n1 + 255) / 256), 256, 0, stream>>>(Wk, WkT, N_HEADS, D_MODEL, D_HEAD);
    cvt_kernel<<<(unsigned)((n2 + 255) / 256), 256, 0, stream>>>(Wv, WvT, N_HEADS * N_EXPERTS, D_MODEL, D_HEAD);
    cvt_kernel<<<(unsigned)((n2 + 255) / 256), 256, 0, stream>>>(Wo, WoT, N_HEADS * N_EXPERTS, D_HEAD, D_MODEL);
  }

  ln_kernel<<<NTOK / 8, 256, 0, stream>>>(x, lng, lnb, xn);
  gate_kernel<<<NTOK / 8, 256, 0, stream>>>(xn, Wsv, Wso, gv, go);

  qk_proj_kernel<<<dim3(NTOK / 64, N_HEADS, 2), 128, 0, stream>>>(xn, WqT, WkT, qbf, kbf);
  v_proj_kernel<<<dim3(NTOK / 16, N_HEADS), 32, 0, stream>>>(xn, WvT, gv, vtb);
  attn_kernel<<<dim3(SEQ / 16, N_HEADS, BATCH), 32, 0, stream>>>(qbf, kbf, vtb, obf);
  out_proj_kernel<<<dim3(NTOK / 16, D_MODEL / 128), 32, 0, stream>>>(obf, WoT, go, x, out);
}
